// SpatialFusion_22119081574819
// MI455X (gfx1250) — compile-verified, matrix operand order checked by reading, not run on hardware
//
#include <hip/hip_runtime.h>
#include <hip/hip_bf16.h>
#include <math.h>

// ---------------------------------------------------------------------------
// Deformable-DETR encoder forward for MI455X (gfx1250, wave32).
// All dense projections run through v_wmma_f32_16x16x32_f16 (f16 in, f32 acc),
// converting fp32->f16 during software-pipelined LDS staging. Activation
// buffers are padded to the 128-row tile so the GEMM hot path has zero bounds
// checks; epilogues fuse bias / residual / ReLU / value-permute, and the
// q = x + pos add is fused into A-staging of the offset/attention GEMMs.
// ---------------------------------------------------------------------------

#define NTOK   9520              // total tokens across 4 levels
#define BATCH2 2
#define MROWS  (BATCH2 * NTOK)   // 19040 real rows
#define MPAD   19072             // 149 * 128 : padded rows for guard-free GEMM
#define DMODEL 256
#define DFFN   1024

typedef __attribute__((ext_vector_type(16))) _Float16 v16h;
typedef __attribute__((ext_vector_type(8)))  float    v8f;
typedef __attribute__((ext_vector_type(4)))  float    f4;
typedef __attribute__((ext_vector_type(4)))  _Float16 h4;
typedef __attribute__((ext_vector_type(2)))  _Float16 h2;

__device__ __constant__ int c_Hs[4] = {64, 32, 16, 8};
__device__ __constant__ int c_Ws[4] = {112, 56, 28, 14};
__device__ __constant__ int c_St[4] = {0, 7168, 8960, 9408};

__device__ __forceinline__ void decode_token(int n, int& l, int& i, int& j) {
    int loc;
    if (n < 7168)      { l = 0; loc = n; }
    else if (n < 8960) { l = 1; loc = n - 7168; }
    else if (n < 9408) { l = 2; loc = n - 8960; }
    else               { l = 3; loc = n - 9408; }
    int W = c_Ws[l];
    i = loc / W; j = loc - i * W;
}

// ---------------------------------------------------------------------------
// Input gather: src_l (B,256,H,W) -> X (MPAD, 256) row-major (real rows only)
// ---------------------------------------------------------------------------
__global__ __launch_bounds__(256) void k_ingather(
    const float* __restrict__ s0, const float* __restrict__ s1,
    const float* __restrict__ s2, const float* __restrict__ s3,
    float* __restrict__ X)
{
    size_t idx = (size_t)blockIdx.x * 256 + threadIdx.x;
    if (idx >= (size_t)MROWS * DMODEL) return;
    int c = (int)(idx & (DMODEL - 1));
    int m = (int)(idx >> 8);
    int n = m % NTOK, b = m / NTOK;
    int l, i, j; decode_token(n, l, i, j);
    const float* s = (l == 0) ? s0 : (l == 1) ? s1 : (l == 2) ? s2 : s3;
    int H = c_Hs[l], W = c_Ws[l];
    X[idx] = s[(((size_t)b * DMODEL + c) * H + i) * W + j];
}

// ---------------------------------------------------------------------------
// Sine positional encoding + level embedding: pos (N,256)
// ---------------------------------------------------------------------------
__global__ __launch_bounds__(256) void k_pos(
    const float* __restrict__ level_embed, float* __restrict__ pos)
{
    size_t idx = (size_t)blockIdx.x * 256 + threadIdx.x;
    if (idx >= (size_t)NTOK * DMODEL) return;
    int c = (int)(idx & (DMODEL - 1));
    int n = (int)(idx >> 8);
    int l, i, j; decode_token(n, l, i, j);
    int H = c_Hs[l], W = c_Ws[l];
    const float scale = 6.283185307179586f;
    float v;
    int cc;
    if (c < 128) { v = ((float)i + 0.5f) / ((float)H + 1e-6f) * scale; cc = c; }
    else         { v = ((float)j + 0.5f) / ((float)W + 1e-6f) * scale; cc = c - 128; }
    float t = powf(10000.0f, (float)(2 * (cc >> 1)) / 128.0f);
    float arg = v / t;
    float p = (cc & 1) ? cosf(arg) : sinf(arg);
    pos[idx] = p + level_embed[l * DMODEL + c];
}

// ref points (N,2) in (x,y)
__global__ __launch_bounds__(256) void k_ref(float* __restrict__ ref)
{
    int n = blockIdx.x * 256 + threadIdx.x;
    if (n >= NTOK) return;
    int l, i, j; decode_token(n, l, i, j);
    ref[2 * n + 0] = ((float)j + 0.5f) / (float)c_Ws[l];
    ref[2 * n + 1] = ((float)i + 0.5f) / (float)c_Hs[l];
}

// ---------------------------------------------------------------------------
// WMMA GEMM: C[MPAD,Nc] = (A[MPAD,K] (+pos)) * B[K,Nc] + bias (+res) (relu).
// Workgroup tile 128x64, 8 waves, wave = 16x64 slab: one A fragment reused
// across four B fragments -> 4 WMMAs per k-tile.  Software pipeline:
//   prefetch next tile's global data into registers BEFORE the WMMAs,
//   convert+store to the other LDS buffer AFTER the WMMAs,
// so global latency hides under matrix math; one barrier per k-tile.
// A/B fragments are loaded into distinct registers so DS waits overlap.
// OUTMODE 0: row-major store.  OUTMODE 1: scatter to value layout (B,8,N,32).
// ---------------------------------------------------------------------------
template<bool RELU, bool RES, bool ADDPOS, int OUTMODE>
__global__ __launch_bounds__(256) void k_gemm_wmma(
    const float* __restrict__ A, const float* __restrict__ Bw,
    const float* __restrict__ bias, const float* __restrict__ Rsd,
    const float* __restrict__ POSp,
    float* __restrict__ C, int Mreal, int K, int Nc)
{
    __shared__ _Float16 As[2][128 * 40];   // row-major, pad 8 halves
    __shared__ _Float16 Bs[2][64 * 40];    // transposed: Bs[n][k]

    const int tid = threadIdx.x;
    const int m0  = blockIdx.y * 128;
    const int n0  = blockIdx.x * 64;
    const int w   = tid >> 5, lane = tid & 31;
    const int lg  = lane >> 4, li = lane & 15;
    const int kp2 = (tid >> 4) * 2, n4 = tid & 15;   // B staging coords

    // per-thread staging coordinates (invariant over k)
    size_t aOff[4]; int aLds[4]; size_t pOff[4];
    #pragma unroll
    for (int i = 0; i < 4; ++i) {
        int q = i * 256 + tid;
        int r = q >> 3, c4 = q & 7;
        aOff[i] = (size_t)(m0 + r) * K + c4 * 4;
        aLds[i] = r * 40 + c4 * 4;
        pOff[i] = ADDPOS ? ((size_t)((m0 + r) % NTOK) * DMODEL + c4 * 4) : 0;
    }
    const size_t bOff = (size_t)kp2 * Nc + n0 + n4 * 4;
    const int    bLds = (n4 * 4) * 40 + kp2;

    f4 ra[4], rp[4], rb0, rb1;     // prefetch registers (all live together)
    auto loadStage = [&](int k0) {
        #pragma unroll
        for (int i = 0; i < 4; ++i) {
            ra[i] = *(const f4*)(A + aOff[i] + k0);
            if (ADDPOS) rp[i] = *(const f4*)(POSp + pOff[i] + k0);
        }
        const float* bp = Bw + (size_t)k0 * Nc + bOff;
        rb0 = *(const f4*)bp;
        rb1 = *(const f4*)(bp + Nc);
    };
    auto storeStage = [&](int buf) {
        #pragma unroll
        for (int i = 0; i < 4; ++i) {
            f4 v = ra[i];
            if (ADDPOS) v = v + rp[i];
            h4 h = { (_Float16)v[0], (_Float16)v[1], (_Float16)v[2], (_Float16)v[3] };
            *(h4*)&As[buf][aLds[i]] = h;
        }
        #pragma unroll
        for (int j = 0; j < 4; ++j) {
            h2 h = { (_Float16)rb0[j], (_Float16)rb1[j] };
            *(h2*)&Bs[buf][bLds + j * 40] = h;
        }
    };

    v8f acc[4];
    acc[0] = v8f{}; acc[1] = v8f{}; acc[2] = v8f{}; acc[3] = v8f{};

    const int T = K >> 5;
    loadStage(0);
    storeStage(0);
    for (int t = 0; t < T; ++t) {
        __syncthreads();
        const bool pf = (t + 1 < T);
        if (pf) loadStage((t + 1) << 5);       // global loads issue early

        const _Float16* as = As[t & 1];
        const _Float16* bs = Bs[t & 1];

        // load A + all four B fragments into distinct regs (progressive waits)
        v16h af, bf0, bf1, bf2, bf3;
        const _Float16* pa = &as[(w * 16 + li) * 40 + lg * 8];
        const _Float16* pb = &bs[li * 40 + lg * 8];
        #pragma unroll
        for (int jj = 0; jj < 8; ++jj) {
            af[jj]  = pa[jj];            af[8 + jj]  = pa[16 + jj];
            bf0[jj] = pb[jj];            bf0[8 + jj] = pb[16 + jj];
            bf1[jj] = pb[16 * 40 + jj];  bf1[8 + jj] = pb[16 * 40 + 16 + jj];
            bf2[jj] = pb[32 * 40 + jj];  bf2[8 + jj] = pb[32 * 40 + 16 + jj];
            bf3[jj] = pb[48 * 40 + jj];  bf3[8 + jj] = pb[48 * 40 + 16 + jj];
        }
        acc[0] = __builtin_amdgcn_wmma_f32_16x16x32_f16(
                     false, af, false, bf0, (short)0, acc[0], false, false);
        acc[1] = __builtin_amdgcn_wmma_f32_16x16x32_f16(
                     false, af, false, bf1, (short)0, acc[1], false, false);
        acc[2] = __builtin_amdgcn_wmma_f32_16x16x32_f16(
                     false, af, false, bf2, (short)0, acc[2], false, false);
        acc[3] = __builtin_amdgcn_wmma_f32_16x16x32_f16(
                     false, af, false, bf3, (short)0, acc[3], false, false);

        if (pf) storeStage((t + 1) & 1);       // convert + LDS store late
    }

    // epilogue — C/D layout: vgpr r, lanes 0-15 -> (M=r, N=li); 16-31 -> (M=8+r)
    const int rowb = m0 + w * 16 + lg * 8;
    #pragma unroll
    for (int u = 0; u < 4; ++u) {
        int col = n0 + u * 16 + li;
        float bv = bias[col];
        #pragma unroll
        for (int r = 0; r < 8; ++r) {
            int row = rowb + r;
            float v = acc[u][r] + bv;
            if (RES)  v += Rsd[(size_t)row * Nc + col];
            if (RELU) v = v > 0.0f ? v : 0.0f;
            if (OUTMODE == 0) {
                C[(size_t)row * Nc + col] = v;      // padded buffer: no guard
            } else if (row < Mreal) {
                // value permute: (b*N+n, h*32+d) -> ((b*8+h)*N+n)*32+d
                int b = row / NTOK, n = row - b * NTOK;
                int h = col >> 5,  d = col & 31;
                C[(((size_t)b * 8 + h) * NTOK + n) * 32 + d] = v;
            }
        }
    }
}

// ---------------------------------------------------------------------------
// Softmax over 16 (levels*points) per (b,n,h), in place on logits
// ---------------------------------------------------------------------------
__global__ __launch_bounds__(256) void k_softmax16(float* __restrict__ ATT, int total)
{
    int g = blockIdx.x * 256 + threadIdx.x;
    if (g >= total) return;
    float* p = ATT + (size_t)g * 16;
    float mx = p[0];
    #pragma unroll
    for (int i = 1; i < 16; ++i) mx = fmaxf(mx, p[i]);
    float e[16]; float s = 0.0f;
    #pragma unroll
    for (int i = 0; i < 16; ++i) { e[i] = __expf(p[i] - mx); s += e[i]; }
    float inv = 1.0f / s;
    #pragma unroll
    for (int i = 0; i < 16; ++i) p[i] = e[i] * inv;
}

// ---------------------------------------------------------------------------
// Deformable sampling: one wave per (b,n,h); lane = head-dim element (dh=32).
// Gathers are 128B contiguous per wave; value pre-permuted to (B,8,N,32).
// ---------------------------------------------------------------------------
__global__ __launch_bounds__(256) void k_sample(
    const float* __restrict__ Vp,   // (B,8,N,32)
    const float* __restrict__ OFF,  // (MPAD, 256) = (b,n,h,l,p,2)
    const float* __restrict__ ATT,  // (MPAD, 128) softmaxed, (h, l*4+p)
    const float* __restrict__ ref,  // (N,2)
    float* __restrict__ Out)        // (MPAD, 256)
{
    int gw   = (int)((blockIdx.x * 256 + threadIdx.x) >> 5);
    int lane = threadIdx.x & 31;
    if (gw >= MROWS * 8) return;
    int h = gw & 7;
    int t = gw >> 3;
    int n = t % NTOK, b = t / NTOK;

    float rx = ref[2 * n], ry = ref[2 * n + 1];
    const float* offp  = OFF + (size_t)t * 256 + h * 32;
    const float* attp  = ATT + (size_t)t * 128 + h * 16;
    const float* vbase = Vp  + ((size_t)(b * 8 + h) * NTOK) * 32;

    float acc = 0.0f;
    #pragma unroll
    for (int l = 0; l < 4; ++l) {
        int H = c_Hs[l], W = c_Ws[l], s = c_St[l];
        float fW = (float)W, fH = (float)H;
        #pragma unroll
        for (int p = 0; p < 4; ++p) {
            int q = l * 4 + p;
            float ox = offp[2 * q], oy = offp[2 * q + 1];
            float x  = (rx + ox / fW) * fW - 0.5f;
            float y  = (ry + oy / fH) * fH - 0.5f;
            float x0 = floorf(x), y0 = floorf(y);
            float wx1 = x - x0, wy1 = y - y0;
            int xi = (int)x0, yi = (int)y0;
            float a = attp[q];
            #pragma unroll
            for (int dy = 0; dy < 2; ++dy) {
                int yy = yi + dy;
                if (yy < 0 || yy >= H) continue;
                float wy = dy ? wy1 : (1.0f - wy1);
                #pragma unroll
                for (int dx = 0; dx < 2; ++dx) {
                    int xx = xi + dx;
                    if (xx < 0 || xx >= W) continue;
                    float wgt = wy * (dx ? wx1 : (1.0f - wx1));
                    acc += a * wgt * vbase[(size_t)(s + yy * W + xx) * 32 + lane];
                }
            }
        }
    }
    Out[(size_t)t * 256 + h * 32 + lane] = acc;
}

// ---------------------------------------------------------------------------
// LayerNorm over 256 cols: one wave per row, shfl_xor reductions (wave32)
// ---------------------------------------------------------------------------
__global__ __launch_bounds__(256) void k_layernorm(
    const float* __restrict__ Y, const float* __restrict__ g,
    const float* __restrict__ be, float* __restrict__ Xo, int M)
{
    int row  = blockIdx.x * 8 + (threadIdx.x >> 5);
    int lane = threadIdx.x & 31;
    if (row >= M) return;
    const float* p = Y + (size_t)row * 256;
    float vals[8]; float s = 0.0f;
    #pragma unroll
    for (int i = 0; i < 8; ++i) { vals[i] = p[lane + 32 * i]; s += vals[i]; }
    #pragma unroll
    for (int m = 16; m >= 1; m >>= 1) s += __shfl_xor(s, m, 32);
    float mean = s * (1.0f / 256.0f);
    float vs = 0.0f;
    #pragma unroll
    for (int i = 0; i < 8; ++i) { float d = vals[i] - mean; vs += d * d; }
    #pragma unroll
    for (int m = 16; m >= 1; m >>= 1) vs += __shfl_xor(vs, m, 32);
    float inv = 1.0f / sqrtf(vs * (1.0f / 256.0f) + 1e-5f);
    float* o = Xo + (size_t)row * 256;
    #pragma unroll
    for (int i = 0; i < 8; ++i) {
        int c = lane + 32 * i;
        o[c] = (vals[i] - mean) * inv * g[c] + be[c];
    }
}

// ---------------------------------------------------------------------------
// Output scatter: X (MPAD,256) -> 4 concatenated (B,256,H,W) tensors
// ---------------------------------------------------------------------------
__global__ __launch_bounds__(256) void k_outscatter(
    const float* __restrict__ X, float* __restrict__ out)
{
    size_t idx = (size_t)blockIdx.x * 256 + threadIdx.x;
    if (idx >= (size_t)MROWS * DMODEL) return;
    int c = (int)(idx & (DMODEL - 1));
    int m = (int)(idx >> 8);
    int n = m % NTOK, b = m / NTOK;
    int l, i, j; decode_token(n, l, i, j);
    int H = c_Hs[l], W = c_Ws[l];
    size_t base = (size_t)BATCH2 * DMODEL * c_St[l];   // concat offset of level l
    out[base + (((size_t)b * DMODEL + c) * H + i) * W + j] = X[idx];
}

// ---------------------------------------------------------------------------
// Orchestration
// ---------------------------------------------------------------------------
extern "C" void kernel_launch(void* const* d_in, const int* in_sizes, int n_in,
                              void* d_out, int out_size, void* d_ws, size_t ws_size,
                              hipStream_t stream)
{
    (void)in_sizes; (void)n_in; (void)out_size; (void)ws_size;
    const float* src[4] = {(const float*)d_in[0], (const float*)d_in[1],
                           (const float*)d_in[2], (const float*)d_in[3]};
    // jax pytree flatten order (sorted dict keys): "layers" < "level_embed";
    // per layer sorted: W1,W2,Wa,Wo,Wp,Wv,b1,b2,ba,bf1,bf2,bo,bp,bv,g1,g2
    auto P = [&](int layer, int k) { return (const float*)d_in[4 + layer * 16 + k]; };
    const float* level_embed = (const float*)d_in[4 + 3 * 16];

    float* W = (float*)d_ws;
    const size_t MP = MPAD;
    size_t o = 0;
    float* X   = W + o; o += MP * 256;
    float* Q   = W + o; o += MP * 256;   // } these four are contiguous:
    float* VP  = W + o; o += MP * 256;   // } reused as the MPAD x 1024 FFN
    float* OFF = W + o; o += MP * 256;   // } hidden buffer (all dead by then)
    float* OB  = W + o; o += MP * 256;   // }
    float* ATT = W + o; o += MP * 128;
    float* Y   = W + o; o += MP * 256;
    float* POS = W + o; o += (size_t)NTOK * 256;
    float* REF = W + o; o += (size_t)NTOK * 2;
    float* Hbuf = Q;   // MPAD x 1024 union (Q region now only used as H)

    const int gElem = (MROWS * 256 + 255) / 256;      // element-wise grids
    const int gTokC = (NTOK * 256 + 255) / 256;
    const int gM128 = MPAD / 128;                     // 149 (exact)
    const int gLN   = (MROWS + 7) / 8;
    const int nGrp  = MROWS * 8;                      // (b,n,h) groups
    const int gSamp = (nGrp * 32 + 255) / 256;
    const int gSm   = (nGrp + 255) / 256;

    k_ingather<<<gElem, 256, 0, stream>>>(src[0], src[1], src[2], src[3], X);
    k_pos<<<gTokC, 256, 0, stream>>>(level_embed, POS);
    k_ref<<<(NTOK + 255) / 256, 256, 0, stream>>>(REF);

    for (int l = 0; l < 3; ++l) {
        const float *W1 = P(l, 0), *W2 = P(l, 1), *Wa = P(l, 2), *Wo = P(l, 3);
        const float *Wp = P(l, 4), *Wv = P(l, 5), *b1 = P(l, 6), *b2 = P(l, 7);
        const float *ba = P(l, 8), *bf1 = P(l, 9), *bf2 = P(l, 10), *bo = P(l, 11);
        const float *bp = P(l, 12), *bv = P(l, 13), *g1 = P(l, 14), *g2 = P(l, 15);

        // value = X @ Wv + bv, scattered to (B,8,N,32) (store-guarded)
        k_gemm_wmma<false, false, false, 1><<<dim3(4, gM128), 256, 0, stream>>>(
            X, Wv, bv, nullptr, nullptr, VP, MROWS, 256, 256);
        // offsets = (X + pos) @ Wo + bo   (q-add fused into staging)
        k_gemm_wmma<false, false, true, 0><<<dim3(4, gM128), 256, 0, stream>>>(
            X, Wo, bo, nullptr, POS, OFF, MROWS, 256, 256);
        // attn logits = (X + pos) @ Wa + ba
        k_gemm_wmma<false, false, true, 0><<<dim3(2, gM128), 256, 0, stream>>>(
            X, Wa, ba, nullptr, POS, ATT, MROWS, 256, 128);
        k_softmax16<<<gSm, 256, 0, stream>>>(ATT, nGrp);
        k_sample<<<gSamp, 256, 0, stream>>>(VP, OFF, ATT, REF, OB);
        // Y = out @ Wp + bp + X (residual fused)
        k_gemm_wmma<false, true, false, 0><<<dim3(4, gM128), 256, 0, stream>>>(
            OB, Wp, bp, X, nullptr, Y, MROWS, 256, 256);
        k_layernorm<<<gLN, 256, 0, stream>>>(Y, g1, b1, X, MROWS);
        // H = relu(X @ W1 + bf1)
        k_gemm_wmma<true, false, false, 0><<<dim3(16, gM128), 256, 0, stream>>>(
            X, W1, bf1, nullptr, nullptr, Hbuf, MROWS, 256, 1024);
        // Y = H @ W2 + bf2 + X (residual fused)
        k_gemm_wmma<false, true, false, 0><<<dim3(4, gM128), 256, 0, stream>>>(
            Hbuf, W2, bf2, X, nullptr, Y, MROWS, 1024, 256);
        k_layernorm<<<gLN, 256, 0, stream>>>(Y, g2, b2, X, MROWS);
    }

    k_outscatter<<<gElem, 256, 0, stream>>>(X, (float*)d_out);
}
